// EdgeNetwork_sparse_48000554500609
// MI455X (gfx1250) — matrix operand order repaired and checked by reading, not compiled, not run on hardware
//
#include <hip/hip_runtime.h>
#include <hip/hip_bf16.h>
#include <math.h>

// Problem constants (from reference)
#define NN    10000      // nodes
#define NE    100000     // edges
#define DD    256        // model dim
#define NH    4          // heads
#define CC    64         // per-head dim
#define NCH   4          // adjacency channels
#define HIDD  8          // hidden dim of small MLPs
#define OUTD  4          // edge-MLP output dim

#define MGROUPS 79       // ceil(625 mtiles / 8 waves)

// gfx1250 async global->LDS copy path (guarded: availability differs per toolchain)
#if defined(__has_builtin)
#  if __has_builtin(__builtin_amdgcn_global_load_async_to_lds_b128) && \
      __has_builtin(__builtin_amdgcn_s_wait_asynccnt)
#    define USE_ASYNC_LDS 1
#  endif
#endif
#ifndef USE_ASYNC_LDS
#  define USE_ASYNC_LDS 0
#endif

typedef __attribute__((ext_vector_type(16))) __bf16 bf16x16;
typedef __attribute__((ext_vector_type(8)))  float  floatx8;
typedef __attribute__((ext_vector_type(4)))  int    v4i;

#if USE_ASYNC_LDS
typedef __attribute__((address_space(1))) v4i* gv4i_p;   // global int4*
typedef __attribute__((address_space(3))) v4i* lv4i_p;   // LDS int4*
#endif

struct B32Frag { uint4 lo, hi; };   // 32 bytes == 16 bf16

// round-to-nearest-even f32 -> bf16 (stored as u16; no scalar __bf16 math needed)
__device__ __forceinline__ unsigned short f2bf(float f) {
    unsigned int u = __builtin_bit_cast(unsigned int, f);
    unsigned int rnd = 0x7FFFu + ((u >> 16) & 1u);
    return (unsigned short)((u + rnd) >> 16);
}

__device__ __forceinline__ bf16x16 load_frag16(const unsigned short* p0,
                                               const unsigned short* p1) {
    B32Frag r;
    r.lo = *reinterpret_cast<const uint4*>(p0);   // 8 bf16
    r.hi = *reinterpret_cast<const uint4*>(p1);   // 8 bf16
    return __builtin_bit_cast(bf16x16, r);
}

__device__ __forceinline__ float elu1(float x) {
    return x > 0.0f ? x : (__expf(x) - 1.0f);
}

// ---------------- conversion kernels ----------------

__global__ void cvt_x_kernel(const float* __restrict__ x,
                             unsigned short* __restrict__ xb, int n) {
    int i = blockIdx.x * blockDim.x + threadIdx.x;
    if (i < n) xb[i] = f2bf(x[i]);
}

// pack weights bf16 as [c][w][d][k], w in {q,k,v,skip}
__global__ void cvt_w_kernel(const float* __restrict__ Wq, const float* __restrict__ Wk,
                             const float* __restrict__ Wv, const float* __restrict__ Ws,
                             unsigned short* __restrict__ wb) {
    int i = blockIdx.x * blockDim.x + threadIdx.x;     // 16*DD*DD exact
    int c    = i / (4 * DD * DD);
    int w    = (i / (DD * DD)) & 3;
    int rest = i & (DD * DD - 1);
    const float* src = (w == 0) ? Wq : (w == 1) ? Wk : (w == 2) ? Wv : Ws;
    wb[i] = f2bf(src[c * DD * DD + rest]);
}

// ---------------- fused Q/K/V/Skip GEMM (bf16 WMMA, f32 accum) ----------------
// out[w][n][d] = sum_k x[n][k] * W_w[d][k] + b_w[d]
// Workgroup = 8 waves, fixed (widx, ngroup), 8 consecutive M-tiles.
// B strip (64 rows x 256 K, 32 KB bf16) staged in LDS via async global->LDS
// copies and shared by all 8 waves; each wave computes a 16(M) x 64(N) strip
// with the A-fragment reused over 4 N-tiles.
// blocks = weight(4) * ngroup(4) * MGROUPS(79) = 1264
__global__ void __launch_bounds__(256) qkvs_gemm_kernel(
        const unsigned short* __restrict__ xb,       // [NN, DD] bf16
        const unsigned short* __restrict__ wb_chan,  // [4, DD, DD] bf16 (q,k,v,skip)
        const float* __restrict__ bq, const float* __restrict__ bk,
        const float* __restrict__ bv, const float* __restrict__ bs,
        float* __restrict__ out)                     // [4, NN, DD]
{
    __shared__ unsigned short lB[64 * DD];           // 32 KB

    const int bid    = blockIdx.x;
    const int widx   = bid / (4 * MGROUPS);
    const int rem    = bid % (4 * MGROUPS);
    const int ngroup = rem / MGROUPS;
    const int mgroup = rem % MGROUPS;

    const int wavei = threadIdx.x >> 5;
    const int lane  = threadIdx.x & 31;
    const int mtile = mgroup * 8 + wavei;
    const int m0 = mtile * 16;
    const int n0 = ngroup * 64;

    const unsigned short* W = wb_chan + (size_t)widx * DD * DD;

    // ---- stage B strip: W rows [n0, n0+64) x K[0,256) -> contiguous 32 KB ----
    const unsigned short* gB = W + (size_t)n0 * DD;
#if USE_ASYNC_LDS
#pragma unroll
    for (int it = 0; it < 8; ++it) {
        int idx = (it * 256 + threadIdx.x) * 8;      // 8 bf16 = 16 B per copy
        __builtin_amdgcn_global_load_async_to_lds_b128(
            (gv4i_p)(gB + idx), (lv4i_p)(&lB[idx]), 0, 0);
    }
    __builtin_amdgcn_s_wait_asynccnt(0);
    __syncthreads();
#else
#pragma unroll
    for (int it = 0; it < 8; ++it) {
        int idx = (it * 256 + threadIdx.x) * 8;
        *reinterpret_cast<uint4*>(&lB[idx]) = *reinterpret_cast<const uint4*>(gB + idx);
    }
    __syncthreads();
#endif

    if (mtile >= 625) return;                        // barrier already passed

    const float* bias = (widx == 0) ? bq : (widx == 1) ? bk : (widx == 2) ? bv : bs;

    // Fragment layouts used here (wave32, 16-bit; contiguous K, same order for A and B):
    // A 16x32: row = lane&15, halves = 16 contiguous K at (lane>>4)*16
    // B 32x16: col = lane&15, halves = 16 contiguous K at (lane>>4)*16
    const int arow = lane & 15;
    const int akb  = (lane >> 4) * 16;
    const int bcol = lane & 15;
    const int bkb  = (lane >> 4) * 16;

    floatx8 acc0 = {}, acc1 = {}, acc2 = {}, acc3 = {};

    const unsigned short* xrow = xb + (size_t)(m0 + arow) * DD;
    const unsigned short* l0 = &lB[( 0 + bcol) * DD + bkb];
    const unsigned short* l1 = &lB[(16 + bcol) * DD + bkb];
    const unsigned short* l2 = &lB[(32 + bcol) * DD + bkb];
    const unsigned short* l3 = &lB[(48 + bcol) * DD + bkb];

    for (int k0 = 0; k0 < DD; k0 += 32) {
        bf16x16 a  = load_frag16(xrow + k0 + akb, xrow + k0 + akb + 8);
        bf16x16 b0 = load_frag16(l0 + k0, l0 + k0 + 8);
        bf16x16 b1 = load_frag16(l1 + k0, l1 + k0 + 8);
        bf16x16 b2 = load_frag16(l2 + k0, l2 + k0 + 8);
        bf16x16 b3 = load_frag16(l3 + k0, l3 + k0 + 8);
        acc0 = __builtin_amdgcn_wmma_f32_16x16x32_bf16(false, a, false, b0, (short)0, acc0, false, false);
        acc1 = __builtin_amdgcn_wmma_f32_16x16x32_bf16(false, a, false, b1, (short)0, acc1, false, false);
        acc2 = __builtin_amdgcn_wmma_f32_16x16x32_bf16(false, a, false, b2, (short)0, acc2, false, false);
        acc3 = __builtin_amdgcn_wmma_f32_16x16x32_bf16(false, a, false, b3, (short)0, acc3, false, false);
    }

    // C/D layout: element r of v8f = out[m0 + r + 8*(lane>>4)][n0 + 16*nt + (lane&15)]
    const int mhalf = (lane >> 4) * 8;
    const int col   = lane & 15;
    const float bb0 = bias[n0 +  0 + col];
    const float bb1 = bias[n0 + 16 + col];
    const float bb2 = bias[n0 + 32 + col];
    const float bb3 = bias[n0 + 48 + col];
    float* outp = out + (size_t)widx * NN * DD;
#pragma unroll
    for (int r = 0; r < 8; ++r) {
        float* orow = outp + (size_t)(m0 + mhalf + r) * DD + n0;
        orow[ 0 + col] = acc0[r] + bb0;
        orow[16 + col] = acc1[r] + bb1;
        orow[32 + col] = acc2[r] + bb2;
        orow[48 + col] = acc3[r] + bb3;
    }
}

// ---------------- degree count ----------------
__global__ void count_kernel(const int* __restrict__ dst, float* __restrict__ cnt) {
    int e = blockIdx.x * blockDim.x + threadIdx.x;
    if (e < NE) atomicAdd(&cnt[dst[e]], 1.0f);
}

// ---------------- per-channel edge kernel: one wave32 per edge ----------------
__global__ void __launch_bounds__(256) edge_kernel(
        const float* __restrict__ q, const float* __restrict__ k,
        const float* __restrict__ v,
        const int* __restrict__ src, const int* __restrict__ dst,
        const float* __restrict__ adjc, const float* __restrict__ Wec,
        float* __restrict__ agg, float* __restrict__ alpha, int chan)
{
    const int e = blockIdx.x * 8 + (threadIdx.x >> 5);
    if (e >= NE) return;
    const int lane = threadIdx.x & 31;
    const int d0 = lane * 8;              // each lane owns 8 dims; head = lane>>3

    const int s = src[e];
    const int t = dst[e];
    const float aw = adjc[e];

    alignas(16) float qd[8], qs[8], kd[8], ks[8], vs[8], ea[8];
    const float4* pqd = reinterpret_cast<const float4*>(q + (size_t)t * DD + d0);
    const float4* pqs = reinterpret_cast<const float4*>(q + (size_t)s * DD + d0);
    const float4* pkd = reinterpret_cast<const float4*>(k + (size_t)t * DD + d0);
    const float4* pks = reinterpret_cast<const float4*>(k + (size_t)s * DD + d0);
    const float4* pvs = reinterpret_cast<const float4*>(v + (size_t)s * DD + d0);
    const float4* pwe = reinterpret_cast<const float4*>(Wec + d0);
    *(float4*)&qd[0] = pqd[0]; *(float4*)&qd[4] = pqd[1];
    *(float4*)&qs[0] = pqs[0]; *(float4*)&qs[4] = pqs[1];
    *(float4*)&kd[0] = pkd[0]; *(float4*)&kd[4] = pkd[1];
    *(float4*)&ks[0] = pks[0]; *(float4*)&ks[4] = pks[1];
    *(float4*)&vs[0] = pvs[0]; *(float4*)&vs[4] = pvs[1];
    *(float4*)&ea[0] = pwe[0]; *(float4*)&ea[4] = pwe[1];

    float s1 = 0.0f, s2 = 0.0f;
#pragma unroll
    for (int i = 0; i < 8; ++i) {
        const float eai = aw * ea[i];
        s1 = fmaf(qd[i], ks[i] + eai, s1);       // q[dst] . (k[src]+ea)
        s2 = fmaf(qs[i], kd[i], s2);             // q[src] . k[dst]
        atomicAdd(&agg[(size_t)t * DD + d0 + i], vs[i] * eai);
    }

    // reduce dot within each head (groups of 8 lanes)
#pragma unroll
    for (int m = 1; m <= 4; m <<= 1) {
        s1 += __shfl_xor(s1, m, 32);
        s2 += __shfl_xor(s2, m, 32);
    }
    const float inv_sqrt_c = 0.125f;             // 1/sqrt(64)
    float ap = 0.5f * (tanhf(s1 * inv_sqrt_c) + tanhf(s2 * inv_sqrt_c));
    // mean over the 4 heads
    ap += __shfl_xor(ap, 8, 32);
    ap += __shfl_xor(ap, 16, 32);
    if (lane == 0) alpha[(size_t)e * NCH + chan] = ap * 0.25f;
}

// ---------------- mean + skip -> x_cat slice ----------------
__global__ void combine_kernel(const float* __restrict__ agg,
                               const float* __restrict__ cnt,
                               const float* __restrict__ skip,
                               float* __restrict__ xcat, int chan)
{
    int i = blockIdx.x * blockDim.x + threadIdx.x;   // NN*DD exact
    int n = i >> 8, d = i & 255;
    float c = fmaxf(cnt[n], 1.0f);
    xcat[(size_t)n * (NCH * DD) + chan * DD + d] = agg[i] / c + skip[i];
}

// ---------------- node MLP: 1024 -> 8 (elu) -> 256 (tanh); one wave per node ----
__global__ void __launch_bounds__(256) mc_mlp_kernel(
        const float* __restrict__ xcat,
        const float* __restrict__ W1, const float* __restrict__ b1,
        const float* __restrict__ W2, const float* __restrict__ b2,
        float* __restrict__ out)
{
    const int n = blockIdx.x * 8 + (threadIdx.x >> 5);
    if (n >= NN) return;
    const int lane = threadIdx.x & 31;

    float h[HIDD];
#pragma unroll
    for (int j = 0; j < HIDD; ++j) h[j] = 0.0f;

    const float* xr = xcat + (size_t)n * (NCH * DD);
    for (int idx = lane; idx < NCH * DD; idx += 32) {
        const float xv = xr[idx];
#pragma unroll
        for (int j = 0; j < HIDD; ++j)
            h[j] = fmaf(xv, W1[j * (NCH * DD) + idx], h[j]);
    }
#pragma unroll
    for (int j = 0; j < HIDD; ++j) {
#pragma unroll
        for (int m = 1; m <= 16; m <<= 1) h[j] += __shfl_xor(h[j], m, 32);
        h[j] = elu1(h[j] + b1[j]);
    }
#pragma unroll
    for (int r = 0; r < 8; ++r) {
        const int d = lane + 32 * r;
        float o = b2[d];
#pragma unroll
        for (int j = 0; j < HIDD; ++j) o = fmaf(h[j], W2[d * HIDD + j], o);
        out[(size_t)n * DD + d] = tanhf(o);
    }
}

// ---------------- edge MLP: 8 -> 8 -> 8 -> 4; thread per edge ----------------
__global__ void edge_mlp_kernel(
        const float* __restrict__ alpha, const float* __restrict__ adj,
        const float* __restrict__ W1, const float* __restrict__ b1,
        const float* __restrict__ W2, const float* __restrict__ b2,
        const float* __restrict__ W3, const float* __restrict__ b3,
        float* __restrict__ out)                    // [OUTD, NE]
{
    int e = blockIdx.x * blockDim.x + threadIdx.x;
    if (e >= NE) return;
    float in[8], h1[8], h2[8];
#pragma unroll
    for (int c = 0; c < NCH; ++c) {
        in[c]       = alpha[(size_t)e * NCH + c];
        in[NCH + c] = adj[(size_t)c * NE + e];
    }
#pragma unroll
    for (int j = 0; j < 8; ++j) {
        float s = b1[j];
#pragma unroll
        for (int i = 0; i < 8; ++i) s = fmaf(in[i], W1[j * 8 + i], s);
        h1[j] = elu1(s);
    }
#pragma unroll
    for (int j = 0; j < 8; ++j) {
        float s = b2[j];
#pragma unroll
        for (int i = 0; i < 8; ++i) s = fmaf(h1[i], W2[j * 8 + i], s);
        h2[j] = elu1(s);
    }
#pragma unroll
    for (int o = 0; o < OUTD; ++o) {
        float s = b3[o];
#pragma unroll
        for (int j = 0; j < 8; ++j) s = fmaf(h2[j], W3[o * 8 + j], s);
        out[(size_t)o * NE + e] = s;
    }
}

// ---------------- host side ----------------
extern "C" void kernel_launch(void* const* d_in, const int* in_sizes, int n_in,
                              void* d_out, int out_size, void* d_ws, size_t ws_size,
                              hipStream_t stream) {
    const float* x      = (const float*)d_in[0];
    const int*   ei     = (const int*)  d_in[1];
    const float* adj    = (const float*)d_in[2];
    // d_in[3] = flags (unused by reference)
    const float* Wq     = (const float*)d_in[4];
    const float* bq     = (const float*)d_in[5];
    const float* Wk     = (const float*)d_in[6];
    const float* bk     = (const float*)d_in[7];
    const float* Wv     = (const float*)d_in[8];
    const float* bv     = (const float*)d_in[9];
    const float* We     = (const float*)d_in[10];
    const float* Wskip  = (const float*)d_in[11];
    const float* bskip  = (const float*)d_in[12];
    const float* W_mc1  = (const float*)d_in[13];
    const float* b_mc1  = (const float*)d_in[14];
    const float* W_mc2  = (const float*)d_in[15];
    const float* b_mc2  = (const float*)d_in[16];
    const float* W_m1   = (const float*)d_in[17];
    const float* b_m1   = (const float*)d_in[18];
    const float* W_m2   = (const float*)d_in[19];
    const float* b_m2   = (const float*)d_in[20];
    const float* W_m3   = (const float*)d_in[21];
    const float* b_m3   = (const float*)d_in[22];

    const int* src = ei;
    const int* dst = ei + NE;

    // workspace carve-out (256B aligned)
    char* ws = (char*)d_ws;
    size_t off = 0;
    auto carve = [&](size_t bytes) -> void* {
        void* p = ws + off;
        off = (off + bytes + 255) & ~(size_t)255;
        return p;
    };
    unsigned short* xb    = (unsigned short*)carve((size_t)NN * DD * 2);       // 5.12 MB
    unsigned short* wb    = (unsigned short*)carve((size_t)16 * DD * DD * 2);  // 2.10 MB
    float*          qkvs  = (float*)carve((size_t)4 * NN * DD * 4);            // 40.96 MB
    float*          agg   = (float*)carve((size_t)NN * DD * 4);                // 10.24 MB
    float*          cnt   = (float*)carve((size_t)NN * 4);                     // 40 KB
    float*          alpha = (float*)carve((size_t)NE * NCH * 4);               // 1.6 MB
    float*          xcat  = (float*)carve((size_t)NN * NCH * DD * 4);          // 40.96 MB

    float* x_out   = (float*)d_out;                    // [NN, DD]
    float* mlp_out = (float*)d_out + (size_t)NN * DD;  // [OUTD, NE]

    // 1) bf16 conversions
    cvt_x_kernel<<<(NN * DD) / 256, 256, 0, stream>>>(x, xb, NN * DD);
    cvt_w_kernel<<<(16 * DD * DD) / 256, 256, 0, stream>>>(Wq, Wk, Wv, Wskip, wb);

    // 2) node in-degree (shared across channels)
    (void)hipMemsetAsync(cnt, 0, (size_t)NN * 4, stream);
    count_kernel<<<(NE + 255) / 256, 256, 0, stream>>>(dst, cnt);

    // 3) per-channel: GEMMs -> edge attention/aggregation -> combine
    for (int c = 0; c < NCH; ++c) {
        qkvs_gemm_kernel<<<4 * 4 * MGROUPS, 256, 0, stream>>>(
            xb, wb + (size_t)c * 4 * DD * DD,
            bq + c * DD, bk + c * DD, bv + c * DD, bskip + c * DD, qkvs);
        (void)hipMemsetAsync(agg, 0, (size_t)NN * DD * 4, stream);
        edge_kernel<<<NE / 8, 256, 0, stream>>>(
            qkvs, qkvs + (size_t)NN * DD, qkvs + (size_t)2 * NN * DD,
            src, dst, adj + (size_t)c * NE, We + c * DD, agg, alpha, c);
        combine_kernel<<<(NN * DD) / 256, 256, 0, stream>>>(
            agg, cnt, qkvs + (size_t)3 * NN * DD, xcat, c);
    }

    // 4) node MLP -> x_out
    mc_mlp_kernel<<<NN / 8, 256, 0, stream>>>(xcat, W_mc1, b_mc1, W_mc2, b_mc2, x_out);

    // 5) edge MLP -> mlp_out ([OUTD, NE])
    edge_mlp_kernel<<<(NE + 255) / 256, 256, 0, stream>>>(
        alpha, adj, W_m1, b_m1, W_m2, b_m2, W_m3, b_m3, mlp_out);
}
